// XSAGQABlock_31988916420611
// MI455X (gfx1250) — matrix operand order repaired and checked by reading, not compiled, not run on hardware
//
#include <hip/hip_runtime.h>

// ---------------------------------------------------------------------------
// XSAGQA transformer block for MI455X (gfx1250, wave32, WMMA).
// All matmuls run through v_wmma_f32_16x16x32_bf16 (f32 accum).
// GEMMs are templated on (N, K) so strides fold into store immediates.
// ---------------------------------------------------------------------------

#define B_   2
#define S_   2048
#define D_   2048
#define H_   16
#define KV_  4
#define HD_  128
#define FFN_ 5632
#define MROWS (B_*S_)

typedef __bf16 v16bf __attribute__((ext_vector_type(16)));
typedef float  v8f   __attribute__((ext_vector_type(8)));

union BFrag { v16bf v; uint4 q[2]; unsigned short u[16]; };

__device__ __forceinline__ unsigned short f2bf(float f) {
    unsigned int u = __float_as_uint(f);
    u += 0x7FFFu + ((u >> 16) & 1u);           // round-to-nearest-even
    return (unsigned short)(u >> 16);
}

__device__ __forceinline__ void wave_lds_fence() {
#if __has_builtin(__builtin_amdgcn_s_wait_dscnt)
    asm volatile("" ::: "memory");
    __builtin_amdgcn_s_wait_dscnt(0);
    asm volatile("" ::: "memory");
#else
    asm volatile("s_wait_dscnt 0" ::: "memory");
#endif
}

// ---------------------------------------------------------------------------
// RMSNorm (wave per row) -> bf16 output
// ---------------------------------------------------------------------------
__global__ __launch_bounds__(256)
void rmsnorm_bf16(const float* __restrict__ x, const float* __restrict__ w,
                  unsigned short* __restrict__ out, int D) {
    const int wave = threadIdx.x >> 5, lane = threadIdx.x & 31;
    const size_t row = (size_t)blockIdx.x * 8 + wave;
    const float* src = x + row * D;
    float ss = 0.f;
    for (int d = lane; d < D; d += 32) { float v = src[d]; ss += v * v; }
    ss += __shfl_xor(ss, 16); ss += __shfl_xor(ss, 8);
    ss += __shfl_xor(ss, 4);  ss += __shfl_xor(ss, 2); ss += __shfl_xor(ss, 1);
    const float rn = rsqrtf(ss / (float)D + 1e-6f);
    unsigned short* dst = out + row * D;
    for (int d = lane; d < D; d += 32) dst[d] = f2bf(src[d] * rn * w[d]);
}

// ---------------------------------------------------------------------------
// WMMA GEMM: C[f32, MROWSxN] = A[bf16, MROWSxK] * Bw[f32, KxN] (Bw converted
// to bf16 while staging into LDS, transposed to [n][k]). Block tile 128x128,
// K-step 32, 8 waves (4x2), each wave 32x64 via 2x4 v_wmma_f32_16x16x32_bf16
// accumulators. N multiple of 128; K multiple of 32. N,K compile-time so all
// strides fold into immediate offsets.
// ---------------------------------------------------------------------------
template <int N, int K>
__global__ __launch_bounds__(256)
void gemm_bf16f32(const unsigned short* __restrict__ A,
                  const float* __restrict__ Bw,
                  float* __restrict__ C) {
    __shared__ __align__(16) unsigned short As[128 * 32];   // [m][k]
    __shared__ __align__(16) unsigned short Bs[128 * 32];   // [n][k] (transposed)
    const int tid = threadIdx.x;
    const int wave = tid >> 5, lane = tid & 31;
    const int bm = blockIdx.y * 128;
    const int bn = blockIdx.x * 128;
    const int wm = (wave & 3) * 32;
    const int wn = (wave >> 2) * 64;
    const v8f zero = {0.f,0.f,0.f,0.f,0.f,0.f,0.f,0.f};
    v8f acc[2][4];
#pragma unroll
    for (int i = 0; i < 2; ++i)
#pragma unroll
        for (int j = 0; j < 4; ++j) acc[i][j] = zero;

    const int ar = tid >> 1, acb = (tid & 1) * 16;   // A tile: 2 thr/row, 16 elems each
    const int bk = tid >> 3, bnb = (tid & 7) * 16;   // B tile: 8 thr/row, 16 elems each
    const int rsel = lane & 15;
    const int ka  = (lane >> 4) * 8;
    const int kb2 = (lane >> 4) * 16;
    const unsigned short* Arow = A + (size_t)(bm + ar) * K + acb;
    const float*          Brow = Bw + (size_t)bk * N + bn + bnb;

    for (int k0 = 0; k0 < K; k0 += 32) {
        {   // A: 128x32 bf16, straight copy
            const uint4* src = (const uint4*)(Arow + k0);
            uint4 x0 = src[0], x1 = src[1];
            *(uint4*)(As + ar * 32 + acb)     = x0;
            *(uint4*)(As + ar * 32 + acb + 8) = x1;
        }
        {   // B: 32x128 f32 -> bf16, store transposed [n][k]
            const float* src = Brow + (size_t)k0 * N;
#pragma unroll
            for (int i = 0; i < 16; ++i)
                Bs[(bnb + i) * 32 + bk] = f2bf(src[i]);
        }
        __syncthreads();
        BFrag af[2], bf[4];
#pragma unroll
        for (int i = 0; i < 2; ++i) {
            const int row = wm + i * 16 + rsel;
            af[i].q[0] = *(const uint4*)(As + row * 32 + ka);
            af[i].q[1] = *(const uint4*)(As + row * 32 + 16 + ka);
        }
#pragma unroll
        for (int j = 0; j < 4; ++j) {
            const int col = wn + j * 16 + rsel;
            bf[j].q[0] = *(const uint4*)(Bs + col * 32 + ka);
            bf[j].q[1] = *(const uint4*)(Bs + col * 32 + 16 + ka);
        }
#pragma unroll
        for (int i = 0; i < 2; ++i)
#pragma unroll
            for (int j = 0; j < 4; ++j)
                acc[i][j] = __builtin_amdgcn_wmma_f32_16x16x32_bf16(
                    false, af[i].v, false, bf[j].v, (short)0, acc[i][j], false, false);
        __syncthreads();
    }
    // Epilogue: one base per 16-row tile; v*N and j*16 are constant offsets
    // that fold into the global_store immediate field.
    const int nsel = lane & 15, hsel = lane >> 4;
#pragma unroll
    for (int i = 0; i < 2; ++i) {
        float* base = C + (size_t)(bm + wm + i * 16 + 8 * hsel) * N
                        + (bn + wn + nsel);
#pragma unroll
        for (int j = 0; j < 4; ++j)
#pragma unroll
            for (int v = 0; v < 8; ++v)
                base[(size_t)v * N + j * 16] = acc[i][j][v];
    }
}

// ---------------------------------------------------------------------------
// Per-head RMSNorm + RoPE; (B,S,NH,HD) f32 -> (B,NH,S,HD) bf16. Wave/(b,s,h).
// ---------------------------------------------------------------------------
__global__ __launch_bounds__(256)
void qk_prep(const float* __restrict__ in, const float* __restrict__ nw,
             const float* __restrict__ freqs, unsigned short* __restrict__ out,
             int NH) {
    const int wave = threadIdx.x >> 5, lane = threadIdx.x & 31;
    const size_t task = (size_t)blockIdx.x * 8 + wave;
    const int hh = (int)(task % NH);
    const int s  = (int)((task / NH) % S_);
    const int b  = (int)(task / ((size_t)NH * S_));
    const float* src = in + (((size_t)b * S_ + s) * NH + hh) * HD_;
    const int d0 = lane * 4;
    float v0 = src[d0], v1 = src[d0+1], v2 = src[d0+2], v3 = src[d0+3];
    float ss = v0*v0 + v1*v1 + v2*v2 + v3*v3;
    ss += __shfl_xor(ss, 16); ss += __shfl_xor(ss, 8);
    ss += __shfl_xor(ss, 4);  ss += __shfl_xor(ss, 2); ss += __shfl_xor(ss, 1);
    const float rn = rsqrtf(ss * (1.0f / HD_) + 1e-6f);
    const float t0 = v0 * rn * nw[d0],     t1 = v1 * rn * nw[d0 + 1];
    const float t2 = v2 * rn * nw[d0 + 2], t3 = v3 * rn * nw[d0 + 3];
    const size_t fb = ((size_t)s * (HD_/2) + lane * 2) * 2;
    const float c0 = freqs[fb + 0], s0v = freqs[fb + 1];
    const float c1 = freqs[fb + 2], s1v = freqs[fb + 3];
    unsigned short* dst = out + (((size_t)b * NH + hh) * S_ + s) * HD_ + d0;
    dst[0] = f2bf(t0 * c0 - t1 * s0v);
    dst[1] = f2bf(t0 * s0v + t1 * c0);
    dst[2] = f2bf(t2 * c1 - t3 * s1v);
    dst[3] = f2bf(t2 * s1v + t3 * c1);
}

// (B,S,KV,HD) f32 -> (B,KV,S,HD) bf16
__global__ __launch_bounds__(256)
void v_prep(const float* __restrict__ in, unsigned short* __restrict__ out) {
    const size_t idx = (size_t)blockIdx.x * 256 + threadIdx.x;
    const int d   = (int)(idx & (HD_ - 1));
    const int kvh = (int)((idx >> 7) & (KV_ - 1));
    const int s   = (int)((idx >> 9) & (S_ - 1));
    const int b   = (int)(idx >> 20);
    out[(((size_t)b * KV_ + kvh) * S_ + s) * HD_ + d] = f2bf(in[idx]);
}

// ---------------------------------------------------------------------------
// Flash attention, strict causal (j < i). One wave per 16-row q-tile.
// 32 keys/iter: 8 WMMAs for q.k^T (4 K-chunks x 2 key tiles) + 8 for P.V.
// P is transposed through per-wave LDS (wave-local s_wait_dscnt; NO block
// barriers since waves have divergent trip counts). Output bf16 in (B,S,D).
// ---------------------------------------------------------------------------
__global__ __launch_bounds__(128)
void attn_fwd(const unsigned short* __restrict__ Q,
              const unsigned short* __restrict__ Km,
              const unsigned short* __restrict__ V,
              unsigned short* __restrict__ O) {
    __shared__ __align__(16) unsigned short pbuf[4][16 * 32];   // per-wave P tile
    __shared__ __align__(16) unsigned short vbuf[4][128 * 32];  // per-wave V^T tile
    const int wave = threadIdx.x >> 5, lane = threadIdx.x & 31;
    const int task = blockIdx.x * 4 + wave;
    const int QT = S_ / 16;
    const int qt = task % QT;
    const int h  = (task / QT) % H_;
    const int b  = task / (QT * H_);
    const int i0 = qt * 16;
    const unsigned short* qb = Q + (size_t)(b * H_ + h) * S_ * HD_;
    const unsigned short* kb = Km + (size_t)(b * KV_ + (h >> 2)) * S_ * HD_;
    const unsigned short* vb = V  + (size_t)(b * KV_ + (h >> 2)) * S_ * HD_;
    const int nn = lane & 15, half = lane >> 4;
    unsigned short* pb  = pbuf[wave];
    unsigned short* vbl = vbuf[wave];

    BFrag qa[4];
    {
        const int row = i0 + nn;
        const int ko = half * 16;
#pragma unroll
        for (int c = 0; c < 4; ++c) {
            qa[c].q[0] = *(const uint4*)(qb + (size_t)row * HD_ + c * 32 + ko);
            qa[c].q[1] = *(const uint4*)(qb + (size_t)row * HD_ + c * 32 + ko + 8);
        }
    }

    const v8f zero = {0.f,0.f,0.f,0.f,0.f,0.f,0.f,0.f};
    v8f acc[8];
    float mrun[8], lrun[8], alph[8];
#pragma unroll
    for (int t = 0; t < 8; ++t) acc[t] = zero;
#pragma unroll
    for (int v = 0; v < 8; ++v) { mrun[v] = -3.0e38f; lrun[v] = 0.f; }
    const float scale = 0.08838834764831845f;   // 1/sqrt(128)

    for (int j0 = 0; j0 < i0 + 16; j0 += 32) {
        // ---- scores: two 16x16 tiles over 4 K-chunks --------------------
        v8f s0 = zero, s1 = zero;
#pragma unroll
        for (int c = 0; c < 4; ++c) {
            BFrag kf0, kf1;
            const unsigned short* p0 = kb + (size_t)(j0 + nn) * HD_ + c * 32 + half * 16;
            const unsigned short* p1 = kb + (size_t)(j0 + 16 + nn) * HD_ + c * 32 + half * 16;
            kf0.q[0] = *(const uint4*)(p0);     kf0.q[1] = *(const uint4*)(p0 + 8);
            kf1.q[0] = *(const uint4*)(p1);     kf1.q[1] = *(const uint4*)(p1 + 8);
            s0 = __builtin_amdgcn_wmma_f32_16x16x32_bf16(false, qa[c].v, false, kf0.v,
                                                         (short)0, s0, false, false);
            s1 = __builtin_amdgcn_wmma_f32_16x16x32_bf16(false, qa[c].v, false, kf1.v,
                                                         (short)0, s1, false, false);
        }
        // ---- online softmax + write P (bf16) into LDS -------------------
#pragma unroll
        for (int v = 0; v < 8; ++v) {
            const int irow = i0 + v + 8 * half;
            const float a0 = s0[v] * scale;
            const float a1 = s1[v] * scale;
            const bool m0 = (j0 + nn) < irow;
            const bool m1 = (j0 + 16 + nn) < irow;
            float t = fmaxf(m0 ? a0 : -3.0e38f, m1 ? a1 : -3.0e38f);
            t = fmaxf(t, __shfl_xor(t, 1)); t = fmaxf(t, __shfl_xor(t, 2));
            t = fmaxf(t, __shfl_xor(t, 4)); t = fmaxf(t, __shfl_xor(t, 8));
            const float mnew = fmaxf(mrun[v], t);
            const float al = __expf(mrun[v] - mnew);
            const float p0v = m0 ? __expf(a0 - mnew) : 0.f;
            const float p1v = m1 ? __expf(a1 - mnew) : 0.f;
            float rs = p0v + p1v;
            rs += __shfl_xor(rs, 1); rs += __shfl_xor(rs, 2);
            rs += __shfl_xor(rs, 4); rs += __shfl_xor(rs, 8);
            lrun[v] = lrun[v] * al + rs;
            mrun[v] = mnew;
            alph[v] = al;
            const int prow = v + 8 * half;
            pb[prow * 32 + nn]      = f2bf(p0v);
            pb[prow * 32 + 16 + nn] = f2bf(p1v);
        }
#pragma unroll
        for (int t2 = 0; t2 < 8; ++t2)
#pragma unroll
            for (int v = 0; v < 8; ++v) acc[t2][v] *= alph[v];
        // ---- stage V tile transposed: vbl[hd*32 + kk] -------------------
        {
            const unsigned short* vsrc = vb + (size_t)(j0 + lane) * HD_;
#pragma unroll
            for (int dg = 0; dg < 16; ++dg) {
                uint4 x = *(const uint4*)(vsrc + dg * 8);
                unsigned short tmp[8];
                *(uint4*)tmp = x;
#pragma unroll
                for (int e = 0; e < 8; ++e)
                    vbl[(dg * 8 + e) * 32 + lane] = tmp[e];
            }
        }
        wave_lds_fence();
        // ---- P.V: 8 output tiles of 16 hd ------------------------------
        BFrag pf;
        pf.q[0] = *(const uint4*)(pb + nn * 32 + half * 16);
        pf.q[1] = *(const uint4*)(pb + nn * 32 + half * 16 + 8);
#pragma unroll
        for (int t2 = 0; t2 < 8; ++t2) {
            BFrag vf;
            vf.q[0] = *(const uint4*)(vbl + (t2 * 16 + nn) * 32 + half * 16);
            vf.q[1] = *(const uint4*)(vbl + (t2 * 16 + nn) * 32 + half * 16 + 8);
            acc[t2] = __builtin_amdgcn_wmma_f32_16x16x32_bf16(
                false, pf.v, false, vf.v, (short)0, acc[t2], false, false);
        }
        wave_lds_fence();
    }
    // ---- epilogue: out bf16 at (b, i, h*HD + hd) -------------------------
#pragma unroll
    for (int v = 0; v < 8; ++v) {
        const float inv = lrun[v] > 0.f ? 1.0f / lrun[v] : 0.f;   // row 0 -> zeros
        const int irow = i0 + v + 8 * half;
        unsigned short* orow = O + ((size_t)b * S_ + irow) * D_ + h * HD_;
#pragma unroll
        for (int t2 = 0; t2 < 8; ++t2)
            orow[t2 * 16 + nn] = f2bf(acc[t2][v] * inv);
    }
}

// ---------------------------------------------------------------------------
// Elementwise kernels
// ---------------------------------------------------------------------------
__global__ __launch_bounds__(256)
void vel_res(const float* __restrict__ x, const float* __restrict__ velin,
             const float* __restrict__ proj, const float* __restrict__ logbeta,
             float* __restrict__ velout, float* __restrict__ xnew) {
    const size_t idx = (size_t)blockIdx.x * 256 + threadIdx.x;
    const float beta = 1.0f / (1.0f + __expf(-logbeta[0]));
    float v = beta * velin[idx] + proj[idx];
    v = fminf(8.0f, fmaxf(-8.0f, v));
    velout[idx] = v;
    xnew[idx] = x[idx] + v;
}

__global__ __launch_bounds__(256)
void silu_mul(const float* __restrict__ g1, const float* __restrict__ g3,
              unsigned short* __restrict__ out) {
    const size_t idx = (size_t)blockIdx.x * 256 + threadIdx.x;
    const float a = g1[idx];
    out[idx] = f2bf((a / (1.0f + __expf(-a))) * g3[idx]);
}

__global__ __launch_bounds__(256)
void final_add(const float* __restrict__ xnew, const float* __restrict__ ffn,
               float* __restrict__ out) {
    const size_t idx = (size_t)blockIdx.x * 256 + threadIdx.x;
    out[idx] = xnew[idx] + ffn[idx];
}

// ---------------------------------------------------------------------------
extern "C" void kernel_launch(void* const* d_in, const int* in_sizes, int n_in,
                              void* d_out, int out_size, void* d_ws, size_t ws_size,
                              hipStream_t stream) {
    (void)in_sizes; (void)n_in; (void)out_size; (void)ws_size;
    const float* x       = (const float*)d_in[0];
    const float* velin   = (const float*)d_in[1];
    const float* freqs   = (const float*)d_in[2];
    const float* prew    = (const float*)d_in[3];
    const float* wq      = (const float*)d_in[4];
    const float* wk      = (const float*)d_in[5];
    const float* wv      = (const float*)d_in[6];
    const float* wo      = (const float*)d_in[7];
    const float* qnw     = (const float*)d_in[8];
    const float* knw     = (const float*)d_in[9];
    const float* logbeta = (const float*)d_in[10];
    const float* ffnw    = (const float*)d_in[11];
    const float* w1      = (const float*)d_in[12];
    const float* w3      = (const float*)d_in[13];
    const float* w2      = (const float*)d_in[14];
    float* out_x   = (float*)d_out;
    float* out_vel = (float*)d_out + (size_t)MROWS * D_;

    char* ws = (char*)d_ws;
    size_t off = 0;
    auto alloc = [&](size_t bytes) -> char* {
        char* p = ws + off;
        off += (bytes + 255) & ~(size_t)255;
        return p;
    };
    unsigned short* h_bf  = (unsigned short*)alloc((size_t)MROWS * D_ * 2);
    float*          qf    = (float*)alloc((size_t)MROWS * (H_ * HD_) * 4);
    float*          kf    = (float*)alloc((size_t)MROWS * (KV_ * HD_) * 4);
    float*          vf    = (float*)alloc((size_t)MROWS * (KV_ * HD_) * 4);
    unsigned short* q_bf  = (unsigned short*)alloc((size_t)MROWS * (H_ * HD_) * 2);
    unsigned short* k_bf  = (unsigned short*)alloc((size_t)MROWS * (KV_ * HD_) * 2);
    unsigned short* v_bf  = (unsigned short*)alloc((size_t)MROWS * (KV_ * HD_) * 2);
    unsigned short* a_bf  = (unsigned short*)alloc((size_t)MROWS * D_ * 2);
    float*          xnew  = (float*)alloc((size_t)MROWS * D_ * 4);
    unsigned short* n2_bf = (unsigned short*)alloc((size_t)MROWS * D_ * 2);
    float*          g1    = (float*)alloc((size_t)MROWS * FFN_ * 4);
    float*          g3    = (float*)alloc((size_t)MROWS * FFN_ * 4);
    unsigned short* ff_bf = (unsigned short*)alloc((size_t)MROWS * FFN_ * 2);
    float* proj = qf;   // safe alias: qf fully consumed by qk_prep before wo GEMM
    float* ffno = g1;   // safe alias: g1 fully consumed by silu_mul before w2 GEMM

    const dim3 blk(256);
    // 1) pre-norm -> bf16 h
    rmsnorm_bf16<<<MROWS / 8, blk, 0, stream>>>(x, prew, h_bf, D_);
    // 2-4) QKV projections
    gemm_bf16f32<H_*HD_, D_><<<dim3((H_ * HD_) / 128, MROWS / 128), blk, 0, stream>>>(h_bf, wq, qf);
    gemm_bf16f32<KV_*HD_, D_><<<dim3((KV_ * HD_) / 128, MROWS / 128), blk, 0, stream>>>(h_bf, wk, kf);
    gemm_bf16f32<KV_*HD_, D_><<<dim3((KV_ * HD_) / 128, MROWS / 128), blk, 0, stream>>>(h_bf, wv, vf);
    // 5-7) head RMSNorm + RoPE + layout change
    qk_prep<<<(B_ * S_ * H_) / 8, blk, 0, stream>>>(qf, qnw, freqs, q_bf, H_);
    qk_prep<<<(B_ * S_ * KV_) / 8, blk, 0, stream>>>(kf, knw, freqs, k_bf, KV_);
    v_prep<<<(B_ * S_ * KV_ * HD_) / 256, blk, 0, stream>>>(vf, v_bf);
    // 8) flash attention (wave per 16-row q tile; 4 waves/block)
    attn_fwd<<<(B_ * H_ * (S_ / 16)) / 4, dim3(128), 0, stream>>>(q_bf, k_bf, v_bf, a_bf);
    // 9) output projection
    gemm_bf16f32<D_, H_*HD_><<<dim3(D_ / 128, MROWS / 128), blk, 0, stream>>>(a_bf, wo, proj);
    // 10) velocity update + residual
    vel_res<<<(MROWS * D_) / 256, blk, 0, stream>>>(x, velin, proj, logbeta, out_vel, xnew);
    // 11) ffn norm
    rmsnorm_bf16<<<MROWS / 8, blk, 0, stream>>>(xnew, ffnw, n2_bf, D_);
    // 12-13) gate/up projections
    gemm_bf16f32<FFN_, D_><<<dim3(FFN_ / 128, MROWS / 128), blk, 0, stream>>>(n2_bf, w1, g1);
    gemm_bf16f32<FFN_, D_><<<dim3(FFN_ / 128, MROWS / 128), blk, 0, stream>>>(n2_bf, w3, g3);
    // 14) SwiGLU
    silu_mul<<<(int)(((size_t)MROWS * FFN_) / 256), blk, 0, stream>>>(g1, g3, ff_bf);
    // 15) down projection
    gemm_bf16f32<D_, FFN_><<<dim3(D_ / 128, MROWS / 128), blk, 0, stream>>>(ff_bf, w2, ffno);
    // 16) final residual
    final_add<<<(MROWS * D_) / 256, blk, 0, stream>>>(xnew, ffno, out_x);
}
